// Model_39676907881693
// MI455X (gfx1250) — matrix operand order repaired and checked by reading, not compiled, not run on hardware
//
#include <hip/hip_runtime.h>

typedef __attribute__((ext_vector_type(16))) _Float16 v16h;
typedef __attribute__((ext_vector_type(8)))  _Float16 v8h;
typedef __attribute__((ext_vector_type(4)))  _Float16 v4h;
typedef __attribute__((ext_vector_type(8)))  float    v8f;

#define BATCH  16
#define LSEQ   2048
#define DDIM   128
#define QTILE  64                 // query rows per workgroup (16 per wave x 4 waves)
#define KTILE  64                 // key rows per main-loop iteration
#define NWAVES 4
#define NITER  (LSEQ / KTILE)     // 32
#define ATT_SCALE 0.08838834764831845f   // 1/sqrt(128)

// All-lanes max across a 16-lane DPP row (matches the C-fragment half-wave):
// row_ror:8/4/2/1 rotate-reduce. Pure VALU -- no LDS bpermute traffic.
__device__ __forceinline__ float rowmax16(float x) {
  int t;
  t = __builtin_amdgcn_update_dpp(0, __float_as_int(x), 0x128, 0xf, 0xf, true);
  x = fmaxf(x, __int_as_float(t));
  t = __builtin_amdgcn_update_dpp(0, __float_as_int(x), 0x124, 0xf, 0xf, true);
  x = fmaxf(x, __int_as_float(t));
  t = __builtin_amdgcn_update_dpp(0, __float_as_int(x), 0x122, 0xf, 0xf, true);
  x = fmaxf(x, __int_as_float(t));
  t = __builtin_amdgcn_update_dpp(0, __float_as_int(x), 0x121, 0xf, 0xf, true);
  x = fmaxf(x, __int_as_float(t));
  return x;
}

__global__ __launch_bounds__(128)
void flash_attn_wmma_f16(const float* __restrict__ x1,
                         const float* __restrict__ x2,
                         float* __restrict__ out) {
  __shared__ alignas(32) _Float16 Ks[KTILE * DDIM];          // [k][d] for QK^T
  __shared__ alignas(32) _Float16 Vt[DDIM * KTILE];          // [d][k] for P@V
  __shared__ alignas(32) _Float16 Ps[NWAVES * 16 * KTILE];   // P reshuffle

  const int tid  = threadIdx.x;
  const int lane = tid & 31;
  const int wave = tid >> 5;
  const int hl   = lane >> 4;     // wave half (fragment layouts)
  const int ln   = lane & 15;

  const int b     = blockIdx.y;
  const int qbase = blockIdx.x * QTILE + wave * 16;

  // ---- Q tile (16x128) -> f16 A-fragments in registers --------------------
  v16h aq[4];
  {
    const float* qrow = x1 + ((size_t)b * LSEQ + qbase + ln) * DDIM;
    #pragma unroll
    for (int c = 0; c < 4; ++c) {
      const int d0 = c * 32 + hl * 16;
      const int d1 = c * 32 + hl * 16 + 8;
      #pragma unroll
      for (int j = 0; j < 8; ++j) {
        aq[c][j]     = (_Float16)qrow[d0 + j];
        aq[c][j + 8] = (_Float16)qrow[d1 + j];
      }
    }
  }

  // constant all-ones B fragment: rowsum(P) = P @ ones via the tensor core
  v16h ones;
  #pragma unroll
  for (int j = 0; j < 16; ++j) ones[j] = (_Float16)1.0f;

  v8f acc[8];                     // 16x128 output accumulator
  v8f lacc = {};                  // running softmax denominator (all cols equal)
  #pragma unroll
  for (int nt = 0; nt < 8; ++nt)
    #pragma unroll
    for (int r = 0; r < 8; ++r) acc[nt][r] = 0.0f;

  float m_run[8];
  #pragma unroll
  for (int r = 0; r < 8; ++r) m_run[r] = -__builtin_inff();

  _Float16* pw = &Ps[wave * 16 * KTILE];
  const int dcol  = (tid & 31) * 4;   // staging: 4 d-columns per thread
  const int rrow0 = tid >> 5;         // staging: starting row

  // ================= main loop over 64-key tiles ===========================
  for (int kt = 0; kt < NITER; ++kt) {
    __syncthreads();   // previous iteration's LDS consumers done

    // Stage 64x128 K/V tile (f32 -> f16), dual layout, float4-vectorized.
    {
      const float* kptr = x2 + ((size_t)b * LSEQ + kt * KTILE) * DDIM;
      #pragma unroll 4
      for (int p = 0; p < 16; ++p) {
        const int r = p * 4 + rrow0;
        const float4 v = *(const float4*)(kptr + r * DDIM + dcol);
        v4h h;
        h[0] = (_Float16)v.x; h[1] = (_Float16)v.y;
        h[2] = (_Float16)v.z; h[3] = (_Float16)v.w;
        *(v4h*)&Ks[r * DDIM + dcol] = h;          // row-major copy: one b64 store
        Vt[(dcol + 0) * KTILE + r] = h[0];        // transposed copy
        Vt[(dcol + 1) * KTILE + r] = h[1];
        Vt[(dcol + 2) * KTILE + r] = h[2];
        Vt[(dcol + 3) * KTILE + r] = h[3];
      }
      if (kt + 1 < NITER) {                       // prefetch next tile into L2
        const float* nxt = kptr + KTILE * DDIM;
        __builtin_prefetch(nxt + tid * 64, 0, 1);
        __builtin_prefetch(nxt + tid * 64 + 32, 0, 1);
      }
    }
    __syncthreads();

    // ---- S = Q @ K^T : 4 N-subtiles x 4 K-chunks ---------------------------
    v8f s[4];
    #pragma unroll
    for (int nt = 0; nt < 4; ++nt)
      #pragma unroll
      for (int r = 0; r < 8; ++r) s[nt][r] = 0.0f;
    #pragma unroll
    for (int c = 0; c < 4; ++c) {
      #pragma unroll
      for (int nt = 0; nt < 4; ++nt) {
        const v16h bk = *(const v16h*)&Ks[(nt * 16 + ln) * DDIM + c * 32 + hl * 16];
        s[nt] = __builtin_amdgcn_wmma_f32_16x16x32_f16(false, aq[c], false, bk,
                                                       (short)0, s[nt], false, false);
      }
    }

    // ---- online softmax: DPP row-max, exp, write P straight to LDS --------
    float corr[8];
    bool  needscale = false;
    #pragma unroll
    for (int r = 0; r < 8; ++r) {
      const float a0 = s[0][r] * ATT_SCALE;
      const float a1 = s[1][r] * ATT_SCALE;
      const float a2 = s[2][r] * ATT_SCALE;
      const float a3 = s[3][r] * ATT_SCALE;
      float mx = fmaxf(fmaxf(a0, a1), fmaxf(a2, a3));
      mx = rowmax16(mx);
      const float m_new = fmaxf(m_run[r], mx);
      corr[r]  = __expf(m_run[r] - m_new);        // == 1.0f exactly if max unchanged
      m_run[r] = m_new;
      needscale |= (corr[r] != 1.0f);
      const int row = r + 8 * hl;
      pw[row * KTILE +      ln] = (_Float16)__expf(a0 - m_new);
      pw[row * KTILE + 16 + ln] = (_Float16)__expf(a1 - m_new);
      pw[row * KTILE + 32 + ln] = (_Float16)__expf(a2 - m_new);
      pw[row * KTILE + 48 + ln] = (_Float16)__expf(a3 - m_new);
    }

    // rescale running accumulators only when some row's max actually moved
    // (wave-uniform branch; common path after warm-up skips 72 VALU muls)
    if (__any(needscale)) {
      #pragma unroll
      for (int nt = 0; nt < 8; ++nt)
        #pragma unroll
        for (int r = 0; r < 8; ++r) acc[nt][r] *= corr[r];
      #pragma unroll
      for (int r = 0; r < 8; ++r) lacc[r] *= corr[r];
    }

    __builtin_amdgcn_wave_barrier();  // keep DS store->load program order

    // ---- reload P as A-fragments (two 16x32 chunks) -----------------------
    v16h pa[2];
    #pragma unroll
    for (int ch = 0; ch < 2; ++ch) {
      const v8h lo = *(const v8h*)&pw[ln * KTILE + ch * 32 + hl * 16];
      const v8h hi = *(const v8h*)&pw[ln * KTILE + ch * 32 + hl * 16 + 8];
      #pragma unroll
      for (int j = 0; j < 8; ++j) { pa[ch][j] = lo[j]; pa[ch][j + 8] = hi[j]; }
    }

    // ---- l += rowsum(P) on the tensor core --------------------------------
    lacc = __builtin_amdgcn_wmma_f32_16x16x32_f16(false, pa[0], false, ones,
                                                  (short)0, lacc, false, false);
    lacc = __builtin_amdgcn_wmma_f32_16x16x32_f16(false, pa[1], false, ones,
                                                  (short)0, lacc, false, false);

    // ---- acc += P(16x64) @ V(64x128) --------------------------------------
    #pragma unroll
    for (int nt = 0; nt < 8; ++nt) {
      #pragma unroll
      for (int ch = 0; ch < 2; ++ch) {
        const v16h bv = *(const v16h*)&Vt[(nt * 16 + ln) * KTILE + ch * 32 + hl * 16];
        acc[nt] = __builtin_amdgcn_wmma_f32_16x16x32_f16(false, pa[ch], false, bv,
                                                         (short)0, acc[nt],
                                                         false, false);
      }
    }
  }

  // ---- epilogue: divide by l (from the ones-matmul), store ---------------
  float inv[8];
  #pragma unroll
  for (int r = 0; r < 8; ++r) inv[r] = 1.0f / lacc[r];

  float* orow = out + ((size_t)b * LSEQ + qbase) * DDIM;
  #pragma unroll
  for (int nt = 0; nt < 8; ++nt)
    #pragma unroll
    for (int r = 0; r < 8; ++r) {
      const int row = r + 8 * hl;
      orow[row * DDIM + nt * 16 + ln] = acc[nt][r] * inv[r];
    }
}

extern "C" void kernel_launch(void* const* d_in, const int* in_sizes, int n_in,
                              void* d_out, int out_size, void* d_ws, size_t ws_size,
                              hipStream_t stream) {
  const float* x1 = (const float*)d_in[0];
  const float* x2 = (const float*)d_in[1];
  float* out = (float*)d_out;
  (void)in_sizes; (void)n_in; (void)out_size; (void)d_ws; (void)ws_size;

  dim3 grid(LSEQ / QTILE, BATCH);   // 32 x 16 workgroups
  dim3 block(NWAVES * 32);          // 4 waves (wave32)
  flash_attn_wmma_f16<<<grid, block, 0, stream>>>(x1, x2, out);
}
